// LSTHM1_20504173871325
// MI455X (gfx1250) — compile-verified, matrix operand order checked by reading, not run on hardware
//
#include <hip/hip_runtime.h>
#include <math.h>

typedef __attribute__((ext_vector_type(16))) __bf16 v16bf;
typedef __attribute__((ext_vector_type(8)))  float  v8f;

#define CELLN 1024
#define BROWS 8192
#define MT 128          // M tile (rows of batch) per block
#define NT 64           // N tile (cols of CELL) per block
#define KT 32           // K step (WMMA bf16 K)
#define APAD 40         // padded K stride (elements) in LDS
#define LDW 4096        // row stride of Ww/Uw/Sw

struct bf16x8 { __bf16 v[8]; };   // 16 bytes

// Workgroup barrier that only drains DScnt (LDS visibility), NOT loadcnt.
// __syncthreads() would drain loadcnt too, serializing the global prefetch.
__device__ __forceinline__ void lds_barrier() {
  asm volatile("s_wait_dscnt 0\n\t"
               "s_barrier_signal -1\n\t"
               "s_barrier_wait -1" ::: "memory");
}

__device__ __forceinline__ v16bf frag_load(const __bf16* p) {
  union { v16bf v; bf16x8 h[2]; } u;
  u.h[0] = *(const bf16x8*)(p);
  u.h[1] = *(const bf16x8*)(p + 16);
  return u.v;
}

__device__ __forceinline__ bf16x8 cvt8(float4 a, float4 b) {
  bf16x8 r;
  r.v[0]=(__bf16)a.x; r.v[1]=(__bf16)a.y; r.v[2]=(__bf16)a.z; r.v[3]=(__bf16)a.w;
  r.v[4]=(__bf16)b.x; r.v[5]=(__bf16)b.y; r.v[6]=(__bf16)b.z; r.v[7]=(__bf16)b.w;
  return r;
}

// ---- Split staging: fetch = pure global loads into regs, commit = cvt +
// ds_store (the loadcnt wait lands in commit, one iteration later, fully
// overlapped with the WMMA block). ----

__device__ __forceinline__ void fetch_a(float4* fa, const float* __restrict__ src,
                                        int lda, int m0, int k0, int t) {
  int r = t >> 1, h = (t & 1) * 16;
  const float4* p = (const float4*)(src + (size_t)(m0 + r) * lda + k0 + h);
  fa[0] = p[0]; fa[1] = p[1]; fa[2] = p[2]; fa[3] = p[3];
}

__device__ __forceinline__ void commit_a(__bf16* As, const float4* fa, int t) {
  int r = t >> 1, h = (t & 1) * 16;
  bf16x8* d = (bf16x8*)(As + r * APAD + h);
  d[0] = cvt8(fa[0], fa[1]);
  d[1] = cvt8(fa[2], fa[3]);
}

__device__ __forceinline__ void fetch_bg(float* fb, const float* __restrict__ W,
                                         int k0, int n0, int t) {
  int g = t >> 6, c = t & 63;
  const float* src = W + (size_t)k0 * LDW + g * CELLN + n0 + c;
#pragma unroll
  for (int kk = 0; kk < KT; ++kk) fb[kk] = src[(size_t)kk * LDW];
}

__device__ __forceinline__ void commit_bg(__bf16* Bs, const float* fb, int t) {
  int g = t >> 6, c = t & 63;
  bf16x8* d = (bf16x8*)(Bs + (g * 64 + c) * APAD);
#pragma unroll
  for (int q = 0; q < 4; ++q) {
    bf16x8 r;
#pragma unroll
    for (int j = 0; j < 8; ++j) r.v[j] = (__bf16)fb[q * 8 + j];
    d[q] = r;
  }
}

__device__ __forceinline__ void fetch_bh(float* fb, const float* __restrict__ V,
                                         int k0, int n0, int t) {
  int q = t >> 6, c = t & 63;
  const float* src = V + (size_t)(k0 + q * 8) * CELLN + n0 + c;
#pragma unroll
  for (int kk = 0; kk < 8; ++kk) fb[kk] = src[(size_t)kk * CELLN];
}

__device__ __forceinline__ void commit_bh(__bf16* Bs, const float* fb, int t) {
  int q = t >> 6, c = t & 63;
  bf16x8 r;
#pragma unroll
  for (int j = 0; j < 8; ++j) r.v[j] = (__bf16)fb[j];
  *(bf16x8*)(Bs + c * APAD + q * 8) = r;
}

__device__ __forceinline__ float sigm(float v) { return 1.0f / (1.0f + __expf(-v)); }
__device__ __forceinline__ float tanh_fast(float v) {
  float e = __expf(2.0f * v);
  return (e - 1.0f) / (e + 1.0f);
}

__global__ __launch_bounds__(256) void lsthm_fused(
    const float* __restrict__ x,   const float* __restrict__ ctm,
    const float* __restrict__ htm, const float* __restrict__ ztm,
    const float* __restrict__ spk,
    const float* __restrict__ Ww,  const float* __restrict__ Wb,
    const float* __restrict__ Uw,  const float* __restrict__ Ub,
    const float* __restrict__ Vw,  const float* __restrict__ Vb,
    const float* __restrict__ Sw,  const float* __restrict__ Sb,
    float* __restrict__ out)
{
  // Double-buffered LDS: one barrier per K-step (2-iteration reuse distance).
  __shared__ __bf16 As[2][MT * APAD];       // 2 x 10240 B
  __shared__ __bf16 Bs[2][4 * 64 * APAD];   // 2 x 20480 B

  const int t    = threadIdx.x;
  const int w    = t >> 5;        // wave id (wave32)
  const int lane = t & 31;
  const int n0   = blockIdx.x * NT;
  const int m0   = blockIdx.y * MT;
  const int lrow = lane & 15;
  const int kh   = (lane >> 4) * 8;

  // Wave footprint: 32 rows x 32 cols (B fragments reused by 2 WMMAs each).
  const int mg = w >> 1;          // 0..3 -> rows mg*32 .. +31
  const int ng = w & 1;           // 0..1 -> cols ng*32 .. +31

  v8f acc[2][4][2];               // [m_subtile][gate][n_subtile]
  v8f accH[2][2];
#pragma unroll
  for (int mi = 0; mi < 2; ++mi) {
#pragma unroll
    for (int g = 0; g < 4; ++g)
#pragma unroll
      for (int sn = 0; sn < 2; ++sn) acc[mi][g][sn] = (v8f)0.0f;
#pragma unroll
    for (int sn = 0; sn < 2; ++sn) accH[mi][sn] = (v8f)0.0f;
  }

  const int a0_off = (mg * 32 + lrow) * APAD + kh;
  const int a1_off = (mg * 32 + 16 + lrow) * APAD + kh;
  const int b_off  = (ng * 32 + lrow) * APAD + kh;

  float4 fa[4];                   // A prefetch (raw f32)
  float  fb[KT];                  // B prefetch (raw f32)
  int p = 0;                      // LDS ping-pong index

  // ---- Phase 1: hybrid = ztm @ Vw (K = 1024) ----
  fetch_a(fa, ztm, CELLN, m0, 0, t);
  fetch_bh(fb, Vw, 0, n0, t);
  for (int k0 = 0; k0 < CELLN; k0 += KT) {
    __bf16* Asb = As[p];
    __bf16* Bsb = Bs[p];
    commit_a(Asb, fa, t);
    commit_bh(Bsb, fb, t);
    if (k0 + KT < CELLN) {
      fetch_a(fa, ztm, CELLN, m0, k0 + KT, t);
      fetch_bh(fb, Vw, k0 + KT, n0, t);
    }
    lds_barrier();
    v16bf a0 = frag_load(Asb + a0_off);
    v16bf a1 = frag_load(Asb + a1_off);
#pragma unroll
    for (int sn = 0; sn < 2; ++sn) {
      v16bf b = frag_load(Bsb + b_off + (sn * 16) * APAD);
      accH[0][sn] = __builtin_amdgcn_wmma_f32_16x16x32_bf16(
          false, a0, false, b, (short)0, accH[0][sn], false, false);
      accH[1][sn] = __builtin_amdgcn_wmma_f32_16x16x32_bf16(
          false, a1, false, b, (short)0, accH[1][sn], false, false);
    }
    p ^= 1;
  }

  // ---- Phase 2: sums = [x|htm|spk] @ [Ww;Uw;Sw] (K = 1024+1024+256) ----
  const float* Aseg[3] = { x, htm, spk };
  const float* Bseg[3] = { Ww, Uw, Sw };
  const int    Kseg[3] = { 1024, 1024, 256 };
  const int    Lseg[3] = { 1024, 1024, 256 };

  for (int seg = 0; seg < 3; ++seg) {
    const float* Ap = Aseg[seg];
    const float* Bp = Bseg[seg];
    const int K   = Kseg[seg];
    const int lda = Lseg[seg];
    fetch_a(fa, Ap, lda, m0, 0, t);
    fetch_bg(fb, Bp, 0, n0, t);
    for (int k0 = 0; k0 < K; k0 += KT) {
      __bf16* Asb = As[p];
      __bf16* Bsb = Bs[p];
      commit_a(Asb, fa, t);
      commit_bg(Bsb, fb, t);
      if (k0 + KT < K) {
        fetch_a(fa, Ap, lda, m0, k0 + KT, t);
        fetch_bg(fb, Bp, k0 + KT, n0, t);
      }
      lds_barrier();
      v16bf a0 = frag_load(Asb + a0_off);
      v16bf a1 = frag_load(Asb + a1_off);
#pragma unroll
      for (int g = 0; g < 4; ++g) {
#pragma unroll
        for (int sn = 0; sn < 2; ++sn) {
          v16bf b = frag_load(Bsb + b_off + (g * 64 + sn * 16) * APAD);
          acc[0][g][sn] = __builtin_amdgcn_wmma_f32_16x16x32_bf16(
              false, a0, false, b, (short)0, acc[0][g][sn], false, false);
          acc[1][g][sn] = __builtin_amdgcn_wmma_f32_16x16x32_bf16(
              false, a1, false, b, (short)0, acc[1][g][sn], false, false);
        }
      }
      p ^= 1;
    }
  }

  // ---- Epilogue: gates + cell update ----
  // C/D layout: lane l holds N = l%16; VGPR v holds M = v + 8*(l/16).
  const int hi = lane >> 4;
  const size_t HOFF = (size_t)BROWS * CELLN;

#pragma unroll
  for (int mi = 0; mi < 2; ++mi) {
    const int mbase = m0 + mg * 32 + mi * 16 + hi * 8;
#pragma unroll
    for (int sn = 0; sn < 2; ++sn) {
      const int col = n0 + ng * 32 + sn * 16 + lrow;
      const float bF = Wb[col]             + Ub[col]             + Sb[col];
      const float bI = Wb[CELLN + col]     + Ub[CELLN + col]     + Sb[CELLN + col];
      const float bO = Wb[2 * CELLN + col] + Ub[2 * CELLN + col] + Sb[2 * CELLN + col];
      const float bC = Wb[3 * CELLN + col] + Ub[3 * CELLN + col] + Sb[3 * CELLN + col];
      const float bH = Vb[col];
#pragma unroll
      for (int v = 0; v < 8; ++v) {
        const size_t idx = (size_t)(mbase + v) * CELLN + col;
        const float ct = ctm[idx];
        const float f  = sigm(acc[mi][0][sn][v] + bF);
        const float i  = sigm(acc[mi][1][sn][v] + bI);
        const float o  = sigm(acc[mi][2][sn][v] + bO);
        const float ch = tanh_fast(acc[mi][3][sn][v] + bC);
        const float fh = sigm(accH[mi][sn][v] + bH);
        const float c  = f * ct + i * ch + fh * ct;
        out[idx]        = c;
        out[HOFF + idx] = tanh_fast(c) * o;
      }
    }
  }
}

extern "C" void kernel_launch(void* const* d_in, const int* in_sizes, int n_in,
                              void* d_out, int out_size, void* d_ws, size_t ws_size,
                              hipStream_t stream) {
  const float* x   = (const float*)d_in[0];
  const float* ctm = (const float*)d_in[1];
  const float* htm = (const float*)d_in[2];
  const float* ztm = (const float*)d_in[3];
  const float* spk = (const float*)d_in[4];
  const float* Ww  = (const float*)d_in[5];
  const float* Wb  = (const float*)d_in[6];
  const float* Uw  = (const float*)d_in[7];
  const float* Ub  = (const float*)d_in[8];
  const float* Vw  = (const float*)d_in[9];
  const float* Vb  = (const float*)d_in[10];
  const float* Sw  = (const float*)d_in[11];
  const float* Sb  = (const float*)d_in[12];
  float* out = (float*)d_out;

  dim3 grid(CELLN / NT, BROWS / MT);   // 16 x 64 = 1024 blocks
  lsthm_fused<<<grid, 256, 0, stream>>>(x, ctm, htm, ztm, spk,
                                        Ww, Wb, Uw, Ub, Vw, Vb, Sw, Sb, out);
}
